// GPTJLayer_44951127720220
// MI455X (gfx1250) — matrix operand order repaired and checked by reading, not compiled, not run on hardware
//
#include <hip/hip_runtime.h>
#include <hip/hip_bf16.h>

// ---------------------------------------------------------------------------
// GPT-NeoX-style layer (parallel residual) for MI455X / gfx1250.
// bf16 WMMA (v_wmma_f32_16x16x32_bf16) with fp32 accumulation,
// TDM (tensor_load_to_lds) staging for activation tiles.
// ---------------------------------------------------------------------------

typedef __attribute__((ext_vector_type(16))) __bf16 v16bf;
typedef __attribute__((ext_vector_type(8)))  float  v8f;
typedef __attribute__((ext_vector_type(4)))  unsigned u32x4;
typedef __attribute__((ext_vector_type(8)))  int      i32x8;
typedef __attribute__((ext_vector_type(4)))  int      i32x4;

union Frag { v16bf v; uint4 q[2]; };

static __device__ __forceinline__ v8f wmma_bf16(v16bf a, v16bf b, v8f c) {
  // (neg_a, A, neg_b, B, c_mod, C, reuse_a, reuse_b)
  return __builtin_amdgcn_wmma_f32_16x16x32_bf16(false, a, false, b, (short)0, c,
                                                 false, false);
}

static __device__ __forceinline__ __bf16 f2bf(float f) {
  unsigned u = __builtin_bit_cast(unsigned, f);
  u = (u + 0x7FFFu + ((u >> 16) & 1u)) >> 16;          // round-to-nearest-even
  unsigned short s = (unsigned short)u;
  return __builtin_bit_cast(__bf16, s);
}

// Model dims (fixed by the reference).
constexpr int S   = 2048;
constexpr int D   = 4096;
constexpr int H   = 16;
constexpr int HD  = 256;
constexpr int DFF = 16384;

// ---------------------------------------------------------------------------
// Tensor Data Mover: 2D bf16 tile (tile_w x tile_h) global -> LDS, with LDS
// row padding.  D# bitfields per CDNA5 ISA §8.3-8.5.
//   pad_int code c -> pad every (2<<c) DWORDs; pad_amt code a -> (a+1) DWORDs.
// Issue from ONE wave; follow with s_wait_tensorcnt(0) + barrier.
// ---------------------------------------------------------------------------
static __device__ __forceinline__ void tdm_load_bf16_2d(
    unsigned lds_off, const void* gsrc, unsigned tensor_w, unsigned tile_w,
    unsigned tile_h, unsigned pad_int, unsigned pad_amt, unsigned row_stride) {
  const unsigned long long ga = (unsigned long long)(size_t)gsrc;
  u32x4 g0;
  g0[0] = 1u;                                           // count=1 (valid user D#)
  g0[1] = lds_off;                                      // lds_addr (bytes)
  g0[2] = (unsigned)ga;                                 // global_addr[31:0]
  g0[3] = (unsigned)((ga >> 32) & 0x01FFFFFFu)          // global_addr[56:32]
        | (2u << 30);                                   // type = 2 ("image")
  i32x8 g1;
  g1[0] = (int)((1u << 16)                              // data_size = 1 -> 2B
              | (1u << 20)                              // pad_enable
              | (pad_int << 22) | (pad_amt << 25));
  g1[1] = (int)((tensor_w & 0xFFFFu) << 16);            // tensor_dim0 lo16
  g1[2] = (int)((tensor_w >> 16) | ((tile_h & 0xFFFFu) << 16)); // td0.hi | td1.lo
  g1[3] = (int)(((tile_h >> 16) & 0xFFFFu) | ((tile_w & 0xFFFFu) << 16)); // td1.hi | tile_dim0
  g1[4] = (int)(tile_h & 0xFFFFu);                      // tile_dim1 | tile_dim2=0
  g1[5] = (int)row_stride;                              // tensor_dim0_stride lo32
  g1[6] = 0;                                            // stride0.hi | stride1.lo
  g1[7] = 0;
  i32x4 z4; z4[0] = 0; z4[1] = 0; z4[2] = 0; z4[3] = 0;
  i32x8 z8;
#pragma unroll
  for (int i = 0; i < 8; ++i) z8[i] = 0;
  // 6-arg form (clang-23 / therock headers): g0, g1, g2, g3, extra group, cpol
  __builtin_amdgcn_tensor_load_to_lds(g0, g1, z4, z4, z8, 0);
}

// ---------------------------------------------------------------------------
// Kernel 1: dual LayerNorm: attn_in = LN1(x), mlp_in = LN2(x)  (bf16 outputs)
// ---------------------------------------------------------------------------
__global__ void __launch_bounds__(256, 1)
ln_dual_kernel(const float* __restrict__ x,
               const float* __restrict__ w1, const float* __restrict__ b1,
               const float* __restrict__ w2, const float* __restrict__ b2,
               __bf16* __restrict__ out1, __bf16* __restrict__ out2) {
  const int row = blockIdx.x;
  const float* xr = x + (size_t)row * D;
  float v[16];
  float s = 0.f, ss = 0.f;
#pragma unroll
  for (int i = 0; i < 16; ++i) {
    v[i] = xr[threadIdx.x + i * 256];
    s  += v[i];
    ss += v[i] * v[i];
  }
#pragma unroll
  for (int d = 16; d >= 1; d >>= 1) {
    s  += __shfl_xor(s,  d, 32);
    ss += __shfl_xor(ss, d, 32);
  }
  __shared__ float rs[8], rss[8];
  const int wave = threadIdx.x >> 5, lane = threadIdx.x & 31;
  if (lane == 0) { rs[wave] = s; rss[wave] = ss; }
  __syncthreads();
  if (threadIdx.x == 0) {
    float a = 0.f, b = 0.f;
#pragma unroll
    for (int i = 0; i < 8; ++i) { a += rs[i]; b += rss[i]; }
    rs[0] = a; rss[0] = b;
  }
  __syncthreads();
  const float mean = rs[0] * (1.0f / D);
  const float var  = rss[0] * (1.0f / D) - mean * mean;
  const float inv  = rsqrtf(var + 1e-5f);
#pragma unroll
  for (int i = 0; i < 16; ++i) {
    const int c = threadIdx.x + i * 256;
    const float nrm = (v[i] - mean) * inv;
    out1[(size_t)row * D + c] = f2bf(nrm * w1[c] + b1[c]);
    out2[(size_t)row * D + c] = f2bf(nrm * w2[c] + b2[c]);
  }
}

// ---------------------------------------------------------------------------
// Kernel 2: GEMM  C[S,N] = A_bf16[S,K] * W_f32[K,N] + bias  (compile-time N,K)
//  MODE 0: QKV scatter -> q[h][s][hd], k[h][s][hd], vt[h][hd][s]  (bf16)
//  MODE 1: attn dense  -> f32 attn_out
//  MODE 2: fc1         -> tanh-GELU, bf16
//  MODE 3: fc2         -> f32 out = val + attn_out + hidden
// 128 threads (4 waves), 64x64 tile, K-step 32, double-buffered LDS,
// A tile staged by TDM, W tile converted fp32->bf16 in flight.
// ---------------------------------------------------------------------------
template <int N, int K, int MODE>
__global__ void __launch_bounds__(128, 1)
gemm_bf16_kernel(const __bf16* __restrict__ A,
                 const float*  __restrict__ W,
                 const float*  __restrict__ bias,
                 void* p0, void* p1, void* p2) {
  __shared__ __align__(16) __bf16 lA[2][64][40];   // [m][k] rows, 40-elem stride
  __shared__ __align__(16) __bf16 lB[2][64][40];   // [n][k] (transposed)

  const int t    = threadIdx.x;
  const int lane = t & 31;
  const int wave = t >> 5;
  const int lr   = lane & 15;
  const int lh   = lane >> 4;

  const int m0 = blockIdx.y * 64;
  const int n0 = blockIdx.x * 64;

  v8f acc[4];
#pragma unroll
  for (int i = 0; i < 4; ++i)
#pragma unroll
    for (int e = 0; e < 8; ++e) acc[i][e] = 0.f;

  const int wn = t & 63, wk = (t >> 6) * 16;       // W-stage: n-in-tile, k-base

  // ---- prologue: stage tile 0 ----
  if (wave == 0)
    tdm_load_bf16_2d((unsigned)(size_t)&lA[0][0][0],
                     A + (size_t)m0 * K, (unsigned)K, 32u, 64u,
                     /*pad 16 DW*/3u, /*+4 DW*/3u, (unsigned)K);
  {
    const float* p = W + (size_t)wk * N + n0 + wn;
    union { __bf16 h[16]; uint4 q[2]; } wp16;
#pragma unroll
    for (int i = 0; i < 16; ++i) wp16.h[i] = f2bf(p[(size_t)i * N]);
    *(uint4*)&lB[0][wn][wk]     = wp16.q[0];
    *(uint4*)&lB[0][wn][wk + 8] = wp16.q[1];
  }
  if (wave == 0) __builtin_amdgcn_s_wait_tensorcnt(0);
  __syncthreads();

  const float* wp = W + (size_t)(32 + wk) * N + n0 + wn;  // next-tile pointer

  for (int k0 = 0; k0 < K; k0 += 32) {
    const int cur = (k0 >> 5) & 1, nxt = cur ^ 1;
    const bool has_next = (k0 + 32) < K;

    // issue next tile early: TDM for A, global loads for W
    float wreg[16];
    if (has_next) {
      if (wave == 0)
        tdm_load_bf16_2d((unsigned)(size_t)&lA[nxt][0][0],
                         A + (size_t)m0 * K + (k0 + 32), (unsigned)K, 32u, 64u,
                         3u, 3u, (unsigned)K);
#pragma unroll
      for (int i = 0; i < 16; ++i) wreg[i] = wp[(size_t)i * N];
      wp += (size_t)32 * N;
    }

    // compute current tile: each wave 16(m) x 64(n)
    Frag a;
    a.q[0] = *(const uint4*)&lA[cur][wave * 16 + lr][lh * 8];
    a.q[1] = *(const uint4*)&lA[cur][wave * 16 + lr][16 + lh * 8];
#pragma unroll
    for (int ns = 0; ns < 4; ++ns) {
      Frag b;
      b.q[0] = *(const uint4*)&lB[cur][ns * 16 + lr][lh * 8];
      b.q[1] = *(const uint4*)&lB[cur][ns * 16 + lr][16 + lh * 8];
      acc[ns] = wmma_bf16(a.v, b.v, acc[ns]);
    }

    if (has_next) {
      union { __bf16 h[16]; uint4 q[2]; } wp16;
#pragma unroll
      for (int i = 0; i < 16; ++i) wp16.h[i] = f2bf(wreg[i]);
      *(uint4*)&lB[nxt][wn][wk]     = wp16.q[0];
      *(uint4*)&lB[nxt][wn][wk + 8] = wp16.q[1];
      if (wave == 0) __builtin_amdgcn_s_wait_tensorcnt(0);
      __syncthreads();
    }
  }

  // ---- epilogue ----
#pragma unroll
  for (int ns = 0; ns < 4; ++ns) {
#pragma unroll
    for (int j = 0; j < 8; ++j) {
      const int row = m0 + wave * 16 + j + lh * 8;   // C: M = j + 8*(lane>=16)
      const int col = n0 + ns * 16 + lr;             //    N = lane % 16
      float val = acc[ns][j] + bias[col];
      if (MODE == 0) {
        const int sec = col >> 12, dd = col & 4095;
        const int hh = dd >> 8, hd = dd & 255;
        const __bf16 bv = f2bf(val);
        if (sec == 0)      ((__bf16*)p0)[((size_t)hh * S + row) * HD + hd] = bv;  // Q[h][s][hd]
        else if (sec == 1) ((__bf16*)p1)[((size_t)hh * S + row) * HD + hd] = bv;  // K[h][s][hd]
        else               ((__bf16*)p2)[((size_t)hh * HD + hd) * S + row] = bv;  // V^T[h][hd][s]
      } else if (MODE == 1) {
        ((float*)p0)[(size_t)row * N + col] = val;
      } else if (MODE == 2) {
        const float u = 0.7978845608028654f * (val + 0.044715f * val * val * val);
        ((__bf16*)p0)[(size_t)row * N + col] = f2bf(0.5f * val * (1.f + tanhf(u)));
      } else {
        const size_t idx = (size_t)row * N + col;
        ((float*)p0)[idx] = val + ((const float*)p1)[idx] + ((const float*)p2)[idx];
      }
    }
  }
}

// ---------------------------------------------------------------------------
// Kernel 3: causal flash attention. One wave per (head, 16-query tile).
// Q,K in [h][s][hd] bf16; V in [h][hd][s] bf16; ctx out [s][d] bf16.
// __launch_bounds__(32,1): single wave, max VGPR budget -> no spills.
// ---------------------------------------------------------------------------
__global__ void __launch_bounds__(32, 1)
attn_kernel(const __bf16* __restrict__ Q,
            const __bf16* __restrict__ Kt,
            const __bf16* __restrict__ Vt,
            __bf16* __restrict__ ctx) {
  const int qt = blockIdx.x;          // 0..S/16-1
  const int h  = blockIdx.y;          // 0..H-1
  const int lane = threadIdx.x;
  const int lr = lane & 15, lh = lane >> 4;
  const int q0 = qt * 16;
  const float scale = 0.0625f;        // 1/sqrt(256)

  __shared__ __align__(16) __bf16 lQ[16][264];   // 16 x 256, 264-elem stride
  __shared__ __align__(16) __bf16 lP[16][40];    // 16 x 32 P staging

  // stage Q tile (16 x 256 bf16 = 8 KB) via TDM; pad 128 DW + 4 DW = 264 elems
  tdm_load_bf16_2d((unsigned)(size_t)&lQ[0][0],
                   Q + ((size_t)(h * S + q0)) * HD, (unsigned)HD, 256u, 16u,
                   /*pad 128 DW*/6u, /*+4 DW*/3u, (unsigned)HD);
  __builtin_amdgcn_s_wait_tensorcnt(0);
  __syncthreads();

  v8f o[16];
#pragma unroll
  for (int tl = 0; tl < 16; ++tl)
#pragma unroll
    for (int e = 0; e < 8; ++e) o[tl][e] = 0.f;
  float m[8], l[8];
#pragma unroll
  for (int j = 0; j < 8; ++j) { m[j] = -1e30f; l[j] = 0.f; }

  const int nkb = (q0 + 15) / 32 + 1;            // causal key-block count
  for (int kb = 0; kb < nkb; ++kb) {
    const int kbase = kb * 32;
    v8f s0, s1;
#pragma unroll
    for (int e = 0; e < 8; ++e) { s0[e] = 0.f; s1[e] = 0.f; }

    // scores: Q(16x256) @ K^T(256x32), two 16x16 tiles over 8 hd-chunks
#pragma unroll
    for (int c = 0; c < 8; ++c) {
      Frag a;
      a.q[0] = *(const uint4*)&lQ[lr][c * 32 + lh * 8];
      a.q[1] = *(const uint4*)&lQ[lr][c * 32 + 16 + lh * 8];
      const uint4* kp0 = (const uint4*)(Kt + ((size_t)(h * S + kbase + lr)) * HD + c * 32 + lh * 8);
      Frag b0; b0.q[0] = kp0[0]; b0.q[1] = kp0[2];
      s0 = wmma_bf16(a.v, b0.v, s0);
      const uint4* kp1 = (const uint4*)(Kt + ((size_t)(h * S + kbase + 16 + lr)) * HD + c * 32 + lh * 8);
      Frag b1; b1.q[0] = kp1[0]; b1.q[1] = kp1[2];
      s1 = wmma_bf16(a.v, b1.v, s1);
    }

    // online softmax update
    float alpha[8];
#pragma unroll
    for (int j = 0; j < 8; ++j) {
      const int qrow = q0 + j + lh * 8;
      float x0 = s0[j] * scale; if (kbase + lr > qrow)      x0 = -1e30f;
      float x1 = s1[j] * scale; if (kbase + 16 + lr > qrow) x1 = -1e30f;
      float mb = fmaxf(x0, x1);
#pragma unroll
      for (int d2 = 1; d2 < 16; d2 <<= 1) mb = fmaxf(mb, __shfl_xor(mb, d2, 32));
      const float mn = fmaxf(m[j], mb);
      const float al = __expf(m[j] - mn);
      const float p0 = __expf(x0 - mn);
      const float p1 = __expf(x1 - mn);
      float rsum = p0 + p1;
#pragma unroll
      for (int d2 = 1; d2 < 16; d2 <<= 1) rsum += __shfl_xor(rsum, d2, 32);
      l[j] = l[j] * al + rsum;
      m[j] = mn;
      alpha[j] = al;
      lP[j + lh * 8][lr]      = f2bf(p0);        // C-layout -> A-layout via LDS
      lP[j + lh * 8][lr + 16] = f2bf(p1);
    }
#pragma unroll
    for (int tl = 0; tl < 16; ++tl)
#pragma unroll
      for (int j = 0; j < 8; ++j) o[tl][j] *= alpha[j];
    __syncthreads();

    // P(16x32) @ V(32x256)
    Frag pa;
    pa.q[0] = *(const uint4*)&lP[lr][lh * 8];
    pa.q[1] = *(const uint4*)&lP[lr][16 + lh * 8];
#pragma unroll
    for (int tl = 0; tl < 16; ++tl) {
      const uint4* vp = (const uint4*)(Vt + ((size_t)(h * HD + tl * 16 + lr)) * S + kbase + lh * 8);
      Frag bv; bv.q[0] = vp[0]; bv.q[1] = vp[2];
      o[tl] = wmma_bf16(pa.v, bv.v, o[tl]);
    }
    __syncthreads();
  }

  // finalize: divide by l, write bf16 ctx[s][d]
#pragma unroll
  for (int j = 0; j < 8; ++j) l[j] = 1.f / l[j];
#pragma unroll
  for (int tl = 0; tl < 16; ++tl)
#pragma unroll
    for (int j = 0; j < 8; ++j) {
      const int row = q0 + j + lh * 8;
      const int col = h * HD + tl * 16 + lr;
      ctx[(size_t)row * D + col] = f2bf(o[tl][j] * l[j]);
    }
}

// ---------------------------------------------------------------------------
// Host launcher
// ---------------------------------------------------------------------------
extern "C" void kernel_launch(void* const* d_in, const int* in_sizes, int n_in,
                              void* d_out, int out_size, void* d_ws, size_t ws_size,
                              hipStream_t stream) {
  (void)in_sizes; (void)n_in; (void)out_size; (void)ws_size;
  const float* hidden  = (const float*)d_in[0];
  const float* ln1_w   = (const float*)d_in[1];
  const float* ln1_b   = (const float*)d_in[2];
  const float* ln2_w   = (const float*)d_in[3];
  const float* ln2_b   = (const float*)d_in[4];
  const float* w_qkv   = (const float*)d_in[5];
  const float* b_qkv   = (const float*)d_in[6];
  const float* w_dense = (const float*)d_in[7];
  const float* b_dense = (const float*)d_in[8];
  const float* w_fc1   = (const float*)d_in[9];
  const float* b_fc1   = (const float*)d_in[10];
  const float* w_fc2   = (const float*)d_in[11];
  const float* b_fc2   = (const float*)d_in[12];
  float* out = (float*)d_out;

  char* ws = (char*)d_ws;
  size_t off = 0;
  auto alloc = [&](size_t bytes) { void* p = ws + off; off += bytes; return p; };
  __bf16* attn_in  = (__bf16*)alloc((size_t)S * D * 2);
  __bf16* mlp_in   = (__bf16*)alloc((size_t)S * D * 2);
  __bf16* qbuf     = (__bf16*)alloc((size_t)S * D * 2);   // [h][s][hd]
  __bf16* kbuf     = (__bf16*)alloc((size_t)S * D * 2);   // [h][s][hd]
  __bf16* vtbuf    = (__bf16*)alloc((size_t)S * D * 2);   // [h][hd][s]
  __bf16* ctx      = (__bf16*)alloc((size_t)S * D * 2);   // [s][d]
  __bf16* hbuf     = (__bf16*)alloc((size_t)S * DFF * 2); // [s][dff]
  float*  attn_out = (float*) alloc((size_t)S * D * 4);   // [s][d]

  // 1) dual LayerNorm
  ln_dual_kernel<<<S, 256, 0, stream>>>(hidden, ln1_w, ln1_b, ln2_w, ln2_b,
                                        attn_in, mlp_in);
  // 2) QKV projection with Q/K/V^T scatter epilogue
  gemm_bf16_kernel<3 * D, D, 0><<<dim3(3 * D / 64, S / 64), 128, 0, stream>>>(
      attn_in, w_qkv, b_qkv, qbuf, kbuf, vtbuf);
  // 3) causal flash attention
  attn_kernel<<<dim3(S / 16, H), 32, 0, stream>>>(qbuf, kbuf, vtbuf, ctx);
  // 4) attention output projection
  gemm_bf16_kernel<D, D, 1><<<dim3(D / 64, S / 64), 128, 0, stream>>>(
      ctx, w_dense, b_dense, attn_out, nullptr, nullptr);
  // 5) fc1 + tanh-GELU
  gemm_bf16_kernel<DFF, D, 2><<<dim3(DFF / 64, S / 64), 128, 0, stream>>>(
      mlp_in, w_fc1, b_fc1, hbuf, nullptr, nullptr);
  // 6) fc2 + attn_out + residual -> d_out
  gemm_bf16_kernel<D, DFF, 3><<<dim3(D / 64, S / 64), 128, 0, stream>>>(
      hbuf, w_fc2, b_fc2, out, attn_out, (void*)hidden);
}
